// BaselineFlashAttention3_34376918237961
// MI455X (gfx1250) — compile-verified, matrix operand order checked by reading, not run on hardware
//
#include <hip/hip_runtime.h>
#include <hip/hip_bf16.h>

// ---------------------------------------------------------------------------
// MI455X (gfx1250, wave32) attention block:
//   q/k/v = x @ W{q,k,v}.T       (bf16 WMMA, f32 accum, async-LDS staging)
//   flash-attention [B=2,H=16,S=2048,D=128] (bf16 WMMA + online softmax,
//       K/V double-buffered in LDS via global_load_async_to_lds_b128)
//   out = attn @ Wo.T            (f32 output)
// ---------------------------------------------------------------------------

typedef __bf16 v16bf __attribute__((ext_vector_type(16)));
typedef float  v8f   __attribute__((ext_vector_type(8)));

constexpr int Bb = 2;
constexpr int Ss = 2048;
constexpr int Cc = 2048;
constexpr int Hh = 16;
constexpr int Dd = 128;
constexpr int Mm = Bb * Ss;           // 4096 tokens
constexpr float SCALE = 0.08838834764831845f;  // 1/sqrt(128)

#define S_WAIT_ASYNC(n) asm volatile("s_wait_asynccnt " #n ::: "memory")

// 16-byte async copy: global -> LDS, tracked by ASYNCcnt (one instr per wave).
__device__ __forceinline__ void async_copy_b128(unsigned lds_dst,
                                                const unsigned short* gsrc) {
    asm volatile("global_load_async_to_lds_b128 %0, %1, off"
                 :: "v"(lds_dst), "v"(gsrc) : "memory");
}
__device__ __forceinline__ unsigned lds_off(const void* p) {
    return (unsigned)(unsigned long long)p;   // low 32 bits = LDS segment offset
}

// float -> bf16 bits, round-to-nearest-even
__device__ __forceinline__ unsigned short f2bfbits(float f) {
    unsigned int u;
    __builtin_memcpy(&u, &f, 4);
    u += 0x7FFFu + ((u >> 16) & 1u);
    return (unsigned short)(u >> 16);
}

// Load a 16x32 (MxK) A-fragment or 32x16 (KxN) B-fragment (mirrored layout)
// from a row-major, K-contiguous bf16 matrix (global or LDS).
// Lane L holds row (L&15); lanes 0-15 carry K=0..7 & 16..23,
// lanes 16-31 carry K=8..15 & 24..31 -> two 16B chunks per lane.
__device__ __forceinline__ v16bf load_frag(const unsigned short* p, int ld, int lane) {
    int r  = lane & 15;
    int kk = (lane >> 4) << 3;
    union { v16bf v; uint4 q[2]; } u;
    u.q[0] = *(const uint4*)(p + (size_t)r * ld + kk);
    u.q[1] = *(const uint4*)(p + (size_t)r * ld + kk + 16);
    return u.v;
}

__device__ __forceinline__ v8f wmma_bf16(v16bf a, v16bf b, v8f c) {
    return __builtin_amdgcn_wmma_f32_16x16x32_bf16(
        false, a, false, b, (short)0, c, false, false);
}
// Same op as the immediately preceding WMMA, with the A-operand reuse hint set
// (ISA: legal when current instruction is identical to the previous one).
__device__ __forceinline__ v8f wmma_bf16_ra(v16bf a, v16bf b, v8f c) {
    return __builtin_amdgcn_wmma_f32_16x16x32_bf16(
        false, a, false, b, (short)0, c, true, false);
}

// ------------------------------ fp32 -> bf16 -------------------------------
__global__ __launch_bounds__(256) void cvt_f32_bf16(const float* __restrict__ in,
                                                    unsigned short* __restrict__ out,
                                                    int n) {
    int i = blockIdx.x * blockDim.x + threadIdx.x;
    int stride = gridDim.x * blockDim.x;
    for (; i < n; i += stride) out[i] = f2bfbits(in[i]);
}

// --------------------------- NT GEMM via WMMA ------------------------------
// C[m,n] = sum_k A[m,k]*B[n,k].  Block tile 128x64 (8 waves, each 32x32).
// A/B K-slices (128x32, 64x32) double-buffered in LDS via async copies.
template <bool BF16OUT>
__global__ __launch_bounds__(256) void gemm_nt(const unsigned short* __restrict__ A,
                                               const unsigned short* __restrict__ Bm,
                                               void* __restrict__ Cout,
                                               int N, int K) {
    __shared__ __align__(16) unsigned short asmem[2][128 * 32];
    __shared__ __align__(16) unsigned short bsmem[2][64 * 32];
    int tid = threadIdx.x, lane = tid & 31, wave = tid >> 5;
    int rowB = blockIdx.y * 128, colB = blockIdx.x * 64;
    int wrow = (wave >> 1) * 32, wcol = (wave & 1) * 32;
    int hi = lane >> 4, nn = lane & 15;

    auto stage = [&](int buf, int k) {
        #pragma unroll
        for (int i = 0; i < 2; ++i) {                       // A: 512 x 16B chunks
            int id = tid + i * 256, r = id >> 2, c4 = id & 3;
            async_copy_b128(lds_off(&asmem[buf][r * 32 + c4 * 8]),
                            A + (size_t)(rowB + r) * K + k + c4 * 8);
        }
        {                                                   // B: 256 x 16B chunks
            int r = tid >> 2, c4 = tid & 3;
            async_copy_b128(lds_off(&bsmem[buf][r * 32 + c4 * 8]),
                            Bm + (size_t)(colB + r) * K + k + c4 * 8);
        }
    };

    v8f zero = {};
    v8f acc[2][2];
    acc[0][0] = zero; acc[0][1] = zero; acc[1][0] = zero; acc[1][1] = zero;

    stage(0, 0);
    int buf = 0;
    for (int k = 0; k < K; k += 32) {
        if (k + 32 < K) { stage(buf ^ 1, k + 32); S_WAIT_ASYNC(3); }
        else            { S_WAIT_ASYNC(0); }
        __syncthreads();
        v16bf a0 = load_frag(&asmem[buf][(wrow     ) * 32], 32, lane);
        v16bf a1 = load_frag(&asmem[buf][(wrow + 16) * 32], 32, lane);
        v16bf b0 = load_frag(&bsmem[buf][(wcol     ) * 32], 32, lane);
        v16bf b1 = load_frag(&bsmem[buf][(wcol + 16) * 32], 32, lane);
        acc[0][0] = wmma_bf16   (a0, b0, acc[0][0]);
        acc[0][1] = wmma_bf16_ra(a0, b1, acc[0][1]);
        acc[1][0] = wmma_bf16   (a1, b0, acc[1][0]);
        acc[1][1] = wmma_bf16_ra(a1, b1, acc[1][1]);
        __syncthreads();
        buf ^= 1;
    }

    #pragma unroll
    for (int i = 0; i < 2; ++i)
        #pragma unroll
        for (int j = 0; j < 2; ++j)
            #pragma unroll
            for (int r = 0; r < 8; ++r) {
                int row = rowB + wrow + i * 16 + r + 8 * hi;
                int col = colB + wcol + j * 16 + nn;
                if (BF16OUT)
                    ((unsigned short*)Cout)[(size_t)row * N + col] = f2bfbits(acc[i][j][r]);
                else
                    ((float*)Cout)[(size_t)row * N + col] = acc[i][j][r];
            }
}

// ----------------------- V transpose: [B,S,C] -> [B,H,D,S] -----------------
__global__ __launch_bounds__(256) void transpose_v(const unsigned short* __restrict__ v,
                                                   unsigned short* __restrict__ vt) {
    __shared__ unsigned short tile[32][33];
    int b = blockIdx.z >> 4, h = blockIdx.z & 15;
    int s0 = blockIdx.x * 32, d0 = blockIdx.y * 32;
    int tx = threadIdx.x, ty = threadIdx.y;  // blockDim = (32, 8)

    const unsigned short* src = v + ((size_t)(b * Ss + s0)) * Cc + h * Dd + d0;
    #pragma unroll
    for (int r = 0; r < 32; r += 8)
        tile[ty + r][tx] = src[(size_t)(ty + r) * Cc + tx];   // tile[s][d]
    __syncthreads();
    unsigned short* dst = vt + ((size_t)((b * Hh + h) * Dd + d0)) * Ss + s0;
    #pragma unroll
    for (int r = 0; r < 32; r += 8)
        dst[(size_t)(ty + r) * Ss + tx] = tile[tx][ty + r];   // out[d][s]
}

// ------------------------------ flash attention ----------------------------
// 4 waves/block, one 16-query tile each; all waves share the (b,h) K/V stream.
// K (32x128) and V^T (128x32) blocks double-buffered in LDS via async copies.
__global__ __launch_bounds__(128) void flash_attn(const unsigned short* __restrict__ q,
                                                  const unsigned short* __restrict__ k,
                                                  const unsigned short* __restrict__ vt,
                                                  unsigned short* __restrict__ o) {
    __shared__ __align__(16) unsigned short kbuf[2][32 * 128];
    __shared__ __align__(16) unsigned short vbuf[2][128 * 32];
    __shared__ __align__(16) unsigned short pbuf[4][16 * 32];
    int tid = threadIdx.x, lane = tid & 31, wave = tid >> 5;
    int b = blockIdx.y >> 4, h = blockIdx.y & 15;
    int q0 = (blockIdx.x * 4 + wave) * 16;
    int hi = lane >> 4, nn = lane & 15;

    const unsigned short* qp = q  + ((size_t)(b * Ss + q0)) * Cc + h * Dd;
    const unsigned short* kp = k  + ((size_t)(b * Ss)) * Cc + h * Dd;
    const unsigned short* vp = vt + ((size_t)((b * Hh + h) * Dd)) * Ss;

    auto stage = [&](int buf, int j) {
        #pragma unroll
        for (int i = 0; i < 4; ++i) {                        // K: 512 x 16B chunks
            int id = tid + i * 128, r = id >> 4, c = id & 15;
            async_copy_b128(lds_off(&kbuf[buf][r * 128 + c * 8]),
                            kp + (size_t)(j + r) * Cc + c * 8);
        }
        #pragma unroll
        for (int i = 0; i < 4; ++i) {                        // V^T: 512 x 16B chunks
            int id = tid + i * 128, d = id >> 2, c = id & 3;
            async_copy_b128(lds_off(&vbuf[buf][d * 32 + c * 8]),
                            vp + (size_t)d * Ss + j + c * 8);
        }
    };

    // Q fragments: 16 queries x 128 dims = 4 x (16x32) A-fragments in regs
    v16bf qf[4];
    #pragma unroll
    for (int c = 0; c < 4; ++c) qf[c] = load_frag(qp + c * 32, Cc, lane);

    v8f zero = {};
    v8f oacc[8];
    #pragma unroll
    for (int t = 0; t < 8; ++t) oacc[t] = zero;
    float mrow[8], lrow[8];
    #pragma unroll
    for (int r = 0; r < 8; ++r) { mrow[r] = -1e30f; lrow[r] = 0.f; }

    stage(0, 0);
    int buf = 0;
    for (int j = 0; j < Ss; j += 32) {
        if (j + 32 < Ss) { stage(buf ^ 1, j + 32); S_WAIT_ASYNC(8); }
        else             { S_WAIT_ASYNC(0); }
        __syncthreads();

        // Load ALL K fragments first (distinct regs -> one DS clause, partial
        // dscnt waits), then run the 8 score WMMAs back-to-back.
        v16bf kf[8];
        #pragma unroll
        for (int c = 0; c < 4; ++c) {
            kf[2 * c]     = load_frag(&kbuf[buf][ 0 * 128 + c * 32], 128, lane);
            kf[2 * c + 1] = load_frag(&kbuf[buf][16 * 128 + c * 32], 128, lane);
        }
        v8f s0 = {}, s1 = {};
        #pragma unroll
        for (int c = 0; c < 4; ++c) {
            s0 = wmma_bf16   (qf[c], kf[2 * c],     s0);
            s1 = wmma_bf16_ra(qf[c], kf[2 * c + 1], s1);   // same A as previous
        }

        // online softmax (row statistics live in each 16-lane half)
        #pragma unroll
        for (int r = 0; r < 8; ++r) {
            float a0 = s0[r] * SCALE, a1 = s1[r] * SCALE;
            float mx = fmaxf(a0, a1);
            #pragma unroll
            for (int m = 8; m >= 1; m >>= 1) mx = fmaxf(mx, __shfl_xor(mx, m, 16));
            float mnew = fmaxf(mrow[r], mx);
            float p0 = __expf(a0 - mnew);
            float p1 = __expf(a1 - mnew);
            float corr = __expf(mrow[r] - mnew);
            float rs = p0 + p1;
            #pragma unroll
            for (int m = 8; m >= 1; m >>= 1) rs += __shfl_xor(rs, m, 16);
            lrow[r] = lrow[r] * corr + rs;
            mrow[r] = mnew;
            #pragma unroll
            for (int t = 0; t < 8; ++t) oacc[t][r] *= corr;
            int row = r + 8 * hi;
            pbuf[wave][row * 32 + nn]      = f2bfbits(p0);
            pbuf[wave][row * 32 + 16 + nn] = f2bfbits(p1);
        }
        // per-wave LDS ops are in-order: P store->load needs no barrier
        v16bf pf = load_frag(&pbuf[wave][0], 32, lane);

        // O += P @ V: preload all 8 V fragments, then 8 WMMAs sharing A=pf
        v16bf vf[8];
        #pragma unroll
        for (int t = 0; t < 8; ++t)
            vf[t] = load_frag(&vbuf[buf][(t * 16) * 32], 32, lane);
        oacc[0] = wmma_bf16(pf, vf[0], oacc[0]);
        #pragma unroll
        for (int t = 1; t < 8; ++t)
            oacc[t] = wmma_bf16_ra(pf, vf[t], oacc[t]);
        __syncthreads();
        buf ^= 1;
    }

    // epilogue: O /= l, merge heads back into [B,S,C]
    unsigned short* op = o + ((size_t)(b * Ss + q0)) * Cc + h * Dd;
    #pragma unroll
    for (int r = 0; r < 8; ++r) {
        float inv = 1.f / lrow[r];
        int row = r + 8 * hi;
        #pragma unroll
        for (int t = 0; t < 8; ++t)
            op[(size_t)row * Cc + t * 16 + nn] = f2bfbits(oacc[t][r] * inv);
    }
}

// ------------------------------- launcher ----------------------------------
extern "C" void kernel_launch(void* const* d_in, const int* in_sizes, int n_in,
                              void* d_out, int out_size, void* d_ws, size_t ws_size,
                              hipStream_t stream) {
    (void)in_sizes; (void)n_in; (void)out_size; (void)ws_size;
    const float* x  = (const float*)d_in[0];
    const float* wq = (const float*)d_in[1];
    const float* wk = (const float*)d_in[2];
    const float* wv = (const float*)d_in[3];
    const float* wo = (const float*)d_in[4];

    char* ws = (char*)d_ws;
    size_t off = 0;
    auto carve = [&](size_t bytes) -> unsigned short* {
        unsigned short* p = (unsigned short*)(ws + off);
        off += (bytes + 255) & ~(size_t)255;
        return p;
    };
    unsigned short* xb  = carve((size_t)Mm * Cc * 2);
    unsigned short* wqb = carve((size_t)Cc * Cc * 2);
    unsigned short* wkb = carve((size_t)Cc * Cc * 2);
    unsigned short* wvb = carve((size_t)Cc * Cc * 2);
    unsigned short* wob = carve((size_t)Cc * Cc * 2);
    unsigned short* qb  = carve((size_t)Mm * Cc * 2);
    unsigned short* kb  = carve((size_t)Mm * Cc * 2);
    unsigned short* vb  = carve((size_t)Mm * Cc * 2);
    unsigned short* vtb = carve((size_t)Bb * Hh * Dd * Ss * 2);
    unsigned short* ob  = carve((size_t)Mm * Cc * 2);

    cvt_f32_bf16<<<2048, 256, 0, stream>>>(x,  xb,  Mm * Cc);
    cvt_f32_bf16<<<2048, 256, 0, stream>>>(wq, wqb, Cc * Cc);
    cvt_f32_bf16<<<2048, 256, 0, stream>>>(wk, wkb, Cc * Cc);
    cvt_f32_bf16<<<2048, 256, 0, stream>>>(wv, wvb, Cc * Cc);
    cvt_f32_bf16<<<2048, 256, 0, stream>>>(wo, wob, Cc * Cc);

    dim3 ggrid(Cc / 64, Mm / 128);
    gemm_nt<true><<<ggrid, 256, 0, stream>>>(xb, wqb, qb, Cc, Cc);
    gemm_nt<true><<<ggrid, 256, 0, stream>>>(xb, wkb, kb, Cc, Cc);
    gemm_nt<true><<<ggrid, 256, 0, stream>>>(xb, wvb, vb, Cc, Cc);

    transpose_v<<<dim3(Ss / 32, Dd / 32, Bb * Hh), dim3(32, 8), 0, stream>>>(vb, vtb);

    flash_attn<<<dim3(Ss / 64, Bb * Hh), 128, 0, stream>>>(qb, kb, vtb, ob);

    gemm_nt<false><<<ggrid, 256, 0, stream>>>(ob, wob, d_out, Cc, Cc);
}
